// ConvolutionalAlexNet_83107617177840
// MI455X (gfx1250) — matrix operand order repaired and checked by reading, not compiled, not run on hardware
//
#include <hip/hip_runtime.h>
#include <hip/hip_bf16.h>

typedef _Float16 h16;
typedef __attribute__((ext_vector_type(8)))  _Float16 v8h;
typedef __attribute__((ext_vector_type(16))) _Float16 v16h;
typedef __attribute__((ext_vector_type(8)))  float    v8f;

// ---------------------------------------------------------------------------
// f32 -> f16 elementwise convert
// ---------------------------------------------------------------------------
__global__ void cvt_f32_f16(const float* __restrict__ in, h16* __restrict__ out, int n) {
  int i = blockIdx.x * 256 + threadIdx.x;
  if (i < n) out[i] = (h16)in[i];
}

// Conv weight repack: OIHW fp32 [Cout][K] -> f16 [Cout][Kpad], zero-padded tail
__global__ void repack_w(const float* __restrict__ w, h16* __restrict__ out,
                         int K, int Kpad, int total) {
  int i = blockIdx.x * 256 + threadIdx.x;
  if (i >= total) return;
  int co = i / Kpad;
  int k  = i - co * Kpad;
  out[i] = (k < K) ? (h16)w[(size_t)co * K + k] : (h16)0.f;
}

// 2x2 maxpool on NCHW f16 (input already ReLU'd)
__global__ void maxpool2_f16(const h16* __restrict__ in, h16* __restrict__ out,
                             int C, int Hin, int Win, int total) {
  int i = blockIdx.x * 256 + threadIdx.x;
  if (i >= total) return;
  int Wo = Win >> 1, Ho = Hin >> 1;
  int ow = i % Wo; int t = i / Wo;
  int oh = t % Ho; t /= Ho;
  int c  = t % C;  int b = t / C;
  const h16* p = in + (((size_t)(b * C + c)) * Hin + oh * 2) * Win + ow * 2;
  float m0 = fmaxf((float)p[0], (float)p[1]);
  float m1 = fmaxf((float)p[Win], (float)p[Win + 1]);
  out[i] = (h16)fmaxf(m0, m1);
}

// ---------------------------------------------------------------------------
// Implicit-GEMM conv, fully compile-time geometry (H/W powers of two -> all
// index math folds to shifts/masks).  One wave computes a (16*MT)x16 output
// slab: the im2col B-fragment gather is built ONCE per K-step and shared by
// MT weight tiles (MT WMMAs per gather).  Accumulate f32; bias+ReLU fused.
// ---------------------------------------------------------------------------
template<int KS, int PAD, int MT, int CIN, int H, int W, int COUT>
__global__ __launch_bounds__(32)
void conv_wmma(const h16* __restrict__ x, const h16* __restrict__ w,
               const float* __restrict__ bias, h16* __restrict__ out) {
  constexpr int K2   = KS * KS;
  constexpr int K    = CIN * K2;
  constexpr int Kpad = (K + 31) & ~31;
  constexpr int HW   = H * W;

  const int lane = threadIdx.x & 31;
  const int lo   = lane & 15;
  const int grp  = lane >> 4;

  const int mbase = blockIdx.y * (16 * MT);
  const int n     = blockIdx.x * 16 + lo;
  const int b     = n / HW;            // HW = 2^k -> shift
  const int rem   = n % HW;
  const int oh    = rem / W;           // W = 2^k -> shift
  const int ow    = rem % W;

  v8f acc[MT];
#pragma unroll
  for (int t = 0; t < MT; ++t) { v8f z = {}; acc[t] = z; }

  // A row pointer for tile t=0; tile t adds t*16*Kpad
  const h16* wrow = w + (size_t)(mbase + lo) * Kpad + grp * 16;

#pragma unroll 1
  for (int kb = 0; kb < Kpad; kb += 32) {
    // B: 32x16 f16 fragment — bfrag[i] = im2col[kb + grp*16 + i][n]
    v16h bfrag;
    const int k0 = kb + grp * 16;
#pragma unroll
    for (int i = 0; i < 16; ++i) {
      int k = k0 + i;
      h16 v = (h16)0.f;
      if (K == Kpad || k < K) {        // guard folds away when K%32==0
        int cin = k / K2;
        int r   = k - cin * K2;
        int kh  = r / KS;
        int kw  = r - kh * KS;
        int ih  = oh + kh - PAD;
        int iw  = ow + kw - PAD;
        if (ih >= 0 && ih < H && iw >= 0 && iw < W)
          v = x[((size_t)(b * CIN + cin) * H + ih) * W + iw];
      }
      bfrag[i] = v;
    }

    // MT weight tiles share this B fragment
#pragma unroll
    for (int t = 0; t < MT; ++t) {
      const h16* wr = wrow + (size_t)t * 16 * Kpad + kb;
      v8h a0 = *(const v8h*)(wr);
      v8h a1 = *(const v8h*)(wr + 8);
      v16h afrag = __builtin_shufflevector(a0, a1,
          0,1,2,3,4,5,6,7,8,9,10,11,12,13,14,15);
      acc[t] = __builtin_amdgcn_wmma_f32_16x16x32_f16(false, afrag, false, bfrag,
                                                      (short)0, acc[t], false, false);
    }
  }

  // Epilogue: acc[t][r] -> (M = mbase + t*16 + 8*grp + r, N = lo)
  const int no   = blockIdx.x * 16 + lo;
  const int b2   = no / HW;
  const int rem2 = no % HW;
  const size_t obase = (size_t)b2 * COUT * HW + rem2;
#pragma unroll
  for (int t = 0; t < MT; ++t) {
#pragma unroll
    for (int r = 0; r < 8; ++r) {
      int mo  = mbase + t * 16 + grp * 8 + r;
      float v = acc[t][r] + bias[mo];
      v = v > 0.f ? v : 0.f;
      out[obase + (size_t)mo * HW] = (h16)v;
    }
  }
}

// ---------------------------------------------------------------------------
// FC GEMM: D[N][M] = X[N][K] * W[M][K]^T + bias.  One wave per (16*MT)x16
// tile; the activation (X) fragment is loaded once per K-step and shared by
// MT weight tiles.
// ---------------------------------------------------------------------------
template<bool RELU, bool OUTF32, int MT>
__global__ __launch_bounds__(32)
void fc_wmma(const h16* __restrict__ wmat, const h16* __restrict__ x,
             const float* __restrict__ bias, void* __restrict__ outv,
             int M, int K) {
  const int lane  = threadIdx.x & 31;
  const int lo    = lane & 15;
  const int grp   = lane >> 4;
  const int mbase = blockIdx.y * (16 * MT);
  const int n     = blockIdx.x * 16 + lo;

  v8f acc[MT];
#pragma unroll
  for (int t = 0; t < MT; ++t) { v8f z = {}; acc[t] = z; }

  const h16* wrow = wmat + (size_t)(mbase + lo) * K + grp * 16;
  const h16* xrow = x    + (size_t)n * K + grp * 16;

#pragma unroll 1
  for (int kb = 0; kb < K; kb += 32) {
    v8h b0 = *(const v8h*)(xrow + kb);
    v8h b1 = *(const v8h*)(xrow + kb + 8);
    v16h bfrag = __builtin_shufflevector(b0, b1,
        0,1,2,3,4,5,6,7,8,9,10,11,12,13,14,15);

#pragma unroll
    for (int t = 0; t < MT; ++t) {
      v8h a0 = {}, a1 = {};
      if (mbase + t * 16 + lo < M) {
        const h16* wr = wrow + (size_t)t * 16 * K + kb;
        a0 = *(const v8h*)(wr);
        a1 = *(const v8h*)(wr + 8);
      }
      v16h afrag = __builtin_shufflevector(a0, a1,
          0,1,2,3,4,5,6,7,8,9,10,11,12,13,14,15);
      acc[t] = __builtin_amdgcn_wmma_f32_16x16x32_f16(false, afrag, false, bfrag,
                                                      (short)0, acc[t], false, false);
    }
  }

  const int no = blockIdx.x * 16 + lo;
#pragma unroll
  for (int t = 0; t < MT; ++t) {
#pragma unroll
    for (int r = 0; r < 8; ++r) {
      int mo = mbase + t * 16 + grp * 8 + r;
      if (mo < M) {
        float v = acc[t][r] + bias[mo];
        if (RELU) v = v > 0.f ? v : 0.f;
        if (OUTF32) ((float*)outv)[(size_t)no * M + mo] = v;
        else        ((h16*)outv)[(size_t)no * M + mo]   = (h16)v;
      }
    }
  }
}

// ---------------------------------------------------------------------------
extern "C" void kernel_launch(void* const* d_in, const int* in_sizes, int n_in,
                              void* d_out, int out_size, void* d_ws, size_t ws_size,
                              hipStream_t stream) {
  const float* x   = (const float*)d_in[0];
  const float* w1  = (const float*)d_in[1];  const float* b1 = (const float*)d_in[2];
  const float* w2  = (const float*)d_in[3];  const float* b2 = (const float*)d_in[4];
  const float* w3  = (const float*)d_in[5];  const float* b3 = (const float*)d_in[6];
  const float* w4  = (const float*)d_in[7];  const float* b4 = (const float*)d_in[8];
  const float* w5  = (const float*)d_in[9];  const float* b5 = (const float*)d_in[10];
  const float* fw1 = (const float*)d_in[11]; const float* fb1 = (const float*)d_in[12];
  const float* fw2 = (const float*)d_in[13]; const float* fb2 = (const float*)d_in[14];
  const float* fw3 = (const float*)d_in[15]; const float* fb3 = (const float*)d_in[16];

  char* ws = (char*)d_ws;
  size_t off = 0;
  auto carve = [&](size_t halves) -> h16* {
    h16* p = (h16*)(ws + off);
    off = (off + halves * 2 + 255) & ~(size_t)255;
    return p;
  };

  h16* x16  = carve((size_t)256 * 3 * 32 * 32);       // 786432
  h16* actA = carve((size_t)256 * 64 * 32 * 32);      // max activation footprint
  h16* actB = carve((size_t)256 * 64 * 32 * 32);
  h16* w1p  = carve((size_t)64  * 384);               // K=363 -> Kpad=384
  h16* w2p  = carve((size_t)192 * 3136);
  h16* w3p  = carve((size_t)384 * 4800);
  h16* w4p  = carve((size_t)256 * 3456);
  h16* w5p  = carve((size_t)256 * 2304);
  h16* f1p  = carve((size_t)4096 * 4096);
  h16* f2p  = carve((size_t)4096 * 4096);
  h16* f3p  = carve((size_t)100 * 4096);

  // --- convert / repack ---
  {
    int n = 256 * 3 * 32 * 32;
    cvt_f32_f16<<<(n + 255) / 256, 256, 0, stream>>>(x, x16, n);
  }
  { int t = 64  * 384;  repack_w<<<(t + 255) / 256, 256, 0, stream>>>(w1, w1p, 363,  384,  t); }
  { int t = 192 * 3136; repack_w<<<(t + 255) / 256, 256, 0, stream>>>(w2, w2p, 3136, 3136, t); }
  { int t = 384 * 4800; repack_w<<<(t + 255) / 256, 256, 0, stream>>>(w3, w3p, 4800, 4800, t); }
  { int t = 256 * 3456; repack_w<<<(t + 255) / 256, 256, 0, stream>>>(w4, w4p, 3456, 3456, t); }
  { int t = 256 * 2304; repack_w<<<(t + 255) / 256, 256, 0, stream>>>(w5, w5p, 2304, 2304, t); }
  { int n = 4096 * 4096; cvt_f32_f16<<<(n + 255) / 256, 256, 0, stream>>>(fw1, f1p, n); }
  { int n = 4096 * 4096; cvt_f32_f16<<<(n + 255) / 256, 256, 0, stream>>>(fw2, f2p, n); }
  { int n = 100  * 4096; cvt_f32_f16<<<(n + 255) / 256, 256, 0, stream>>>(fw3, f3p, n); }

  // --- features: <KS,PAD,MT,CIN,H,W,COUT>, grid.y = COUT/(16*MT) ---
  // conv1: 3->64, 32x32, k11 p5   (MT=4 covers all 64 couts)
  conv_wmma<11, 5, 4, 3, 32, 32, 64>
      <<<dim3(256 * 32 * 32 / 16, 1), 32, 0, stream>>>(x16, w1p, b1, actA);
  { int t = 256 * 64 * 16 * 16;
    maxpool2_f16<<<(t + 255) / 256, 256, 0, stream>>>(actA, actB, 64, 32, 32, t); }

  // conv2: 64->192, 16x16, k7 p3  (MT=6)
  conv_wmma<7, 3, 6, 64, 16, 16, 192>
      <<<dim3(256 * 16 * 16 / 16, 2), 32, 0, stream>>>(actB, w2p, b2, actA);
  { int t = 256 * 192 * 8 * 8;
    maxpool2_f16<<<(t + 255) / 256, 256, 0, stream>>>(actA, actB, 192, 16, 16, t); }

  // conv3: 192->384, 8x8, k5 p2   (MT=8)
  conv_wmma<5, 2, 8, 192, 8, 8, 384>
      <<<dim3(256 * 8 * 8 / 16, 3), 32, 0, stream>>>(actB, w3p, b3, actA);
  // conv4: 384->256, 8x8, k3 p1   (MT=8)
  conv_wmma<3, 1, 8, 384, 8, 8, 256>
      <<<dim3(256 * 8 * 8 / 16, 2), 32, 0, stream>>>(actA, w4p, b4, actB);
  // conv5: 256->256, 8x8, k3 p1   (MT=8)
  conv_wmma<3, 1, 8, 256, 8, 8, 256>
      <<<dim3(256 * 8 * 8 / 16, 2), 32, 0, stream>>>(actB, w5p, b5, actA);
  { int t = 256 * 256 * 4 * 4;
    maxpool2_f16<<<(t + 255) / 256, 256, 0, stream>>>(actA, actB, 256, 8, 8, t); }
  // actB now holds [256][4096] f16 (NCHW flatten == reference reshape order)

  // --- classifier (MT=4 -> grid.y covers M/64) ---
  fc_wmma<true,  false, 4><<<dim3(256 / 16, 4096 / 64), 32, 0, stream>>>(
      f1p, actB, fb1, actA, 4096, 4096);
  fc_wmma<true,  false, 4><<<dim3(256 / 16, 4096 / 64), 32, 0, stream>>>(
      f2p, actA, fb2, actB, 4096, 4096);
  fc_wmma<false, true,  4><<<dim3(256 / 16, (100 + 63) / 64), 32, 0, stream>>>(
      f3p, actB, fb3, d_out, 100, 4096);

  (void)in_sizes; (void)n_in; (void)out_size; (void)ws_size;
}
